// SimpleConvolution_82918638617112
// MI455X (gfx1250) — matrix operand order repaired and checked by reading, not compiled, not run on hardware
//
#include <hip/hip_runtime.h>

// ---------------------------------------------------------------------------
// GCN layer for MI455X (gfx1250, wave32, WMMA):
//   H   = X @ W + b            (f16 storage, f32 accumulate via v_wmma)
//   out = (A @ H) / rowsum(A)  (degree fused into the aggregation K-loop)
//
// H is kept TRANSPOSED in workspace: Ht[b][col][node]  (f16)
//  -> WMMA B-fragments are contiguous in LDS (no scalar gathers / v_mov_b16)
//  -> projection epilogue stores one 16B vector per accumulator
// W is pre-transposed to Wt[col][k] so the projection uses the same B path.
//
// d_ws layout (requires ~34 MB):
//   [0 .. 16M halves)  : Ht_f16  (8*512*4096)
//   [16M .. +256K)     : Wt_f16  (512*512, transposed)
// ---------------------------------------------------------------------------

typedef __attribute__((ext_vector_type(16))) _Float16 v16h;
typedef __attribute__((ext_vector_type(8)))  _Float16 v8h;
typedef __attribute__((ext_vector_type(4)))  _Float16 v4h;
typedef __attribute__((ext_vector_type(8)))  float    v8f;

static constexpr int BB = 8;
static constexpr int NN = 4096;
static constexpr int CC = 512;
static constexpr int TM = 32;          // output rows per block
static constexpr int KT = 32;          // K-slab per step
static constexpr int AS_STRIDE = 40;   // halves; A rows 80B -> 16B aligned
static constexpr int BT_STRIDE = 36;   // halves; 18 dwords -> conflict-free b64 access

#define SHUF8(a, b)  __builtin_shufflevector(a, b, 0, 1, 2, 3, 4, 5, 6, 7)
#define SHUF16(a, b) __builtin_shufflevector(a, b, 0, 1, 2, 3, 4, 5, 6, 7, 8, 9, 10, 11, 12, 13, 14, 15)

// ---- W fp32 -> f16, transposed: Wt[c][k] = W[k][c] -------------------------
__global__ void gcn_wcvt(const float* __restrict__ W, _Float16* __restrict__ Wt) {
    int i = blockIdx.x * blockDim.x + threadIdx.x;   // i = c*512 + k
    int c = i >> 9, k = i & 511;
    Wt[i] = (_Float16)W[k * CC + c];                 // coalesced writes
}

// ---- B-slab global->LDS: 32 contiguous halves of one column ---------------
__device__ __forceinline__ void stage_bcol(_Float16* Bt, int col,
                                           const _Float16* __restrict__ gsrc) {
    v8h g0 = *(const v8h*)(gsrc);
    v8h g1 = *(const v8h*)(gsrc + 8);
    v8h g2 = *(const v8h*)(gsrc + 16);
    v8h g3 = *(const v8h*)(gsrc + 24);
    _Float16* dp = Bt + col * BT_STRIDE;
    *(v4h*)(dp +  0) = __builtin_shufflevector(g0, g0, 0, 1, 2, 3);
    *(v4h*)(dp +  4) = __builtin_shufflevector(g0, g0, 4, 5, 6, 7);
    *(v4h*)(dp +  8) = __builtin_shufflevector(g1, g1, 0, 1, 2, 3);
    *(v4h*)(dp + 12) = __builtin_shufflevector(g1, g1, 4, 5, 6, 7);
    *(v4h*)(dp + 16) = __builtin_shufflevector(g2, g2, 0, 1, 2, 3);
    *(v4h*)(dp + 20) = __builtin_shufflevector(g2, g2, 4, 5, 6, 7);
    *(v4h*)(dp + 24) = __builtin_shufflevector(g3, g3, 0, 1, 2, 3);
    *(v4h*)(dp + 28) = __builtin_shufflevector(g3, g3, 4, 5, 6, 7);
}

// ---- B-fragment: contiguous 32B per lane, conflict-free b64 loads ---------
__device__ __forceinline__ v16h load_bfrag(const _Float16* Bt, int col, int kg) {
    const _Float16* bp = Bt + col * BT_STRIDE + kg * 16;
    v4h p0 = *(const v4h*)(bp);
    v4h p1 = *(const v4h*)(bp + 4);
    v4h p2 = *(const v4h*)(bp + 8);
    v4h p3 = *(const v4h*)(bp + 12);
    return SHUF16(SHUF8(p0, p1), SHUF8(p2, p3));     // element e -> K = kg*16+e
}

// ---- WMMA inner step (A-tile in As, B-slab in Bt) -------------------------
// wave computes 2 (M) x 4 (N16) accumulators over columns [wave*64, wave*64+64)
__device__ __forceinline__ void wmma_step(const _Float16* As, const _Float16* Bt,
                                          int lane, int wave, v8f acc[2][4]) {
    const int m0 = lane & 15;   // row-in-tile / col-in-tile
    const int kh = lane >> 4;   // K half-group
    v16h afrag[2];
#pragma unroll
    for (int mt = 0; mt < 2; ++mt) {
        const _Float16* ap = As + (mt * 16 + m0) * AS_STRIDE + kh * 16;
        v8h lo = *(const v8h*)(ap);        // K = 16*kh .. 16*kh+7
        v8h hi = *(const v8h*)(ap + 8);    // K = 16*kh+8 .. 16*kh+15
        afrag[mt] = SHUF16(lo, hi);
    }
#pragma unroll
    for (int nt = 0; nt < 4; ++nt) {
        v16h bfrag = load_bfrag(Bt, wave * 64 + nt * 16 + m0, kh);
#pragma unroll
        for (int mt = 0; mt < 2; ++mt)
            acc[mt][nt] = __builtin_amdgcn_wmma_f32_16x16x32_f16(
                false, afrag[mt], false, bfrag, (short)0, acc[mt][nt], false, false);
    }
}

// ---- projection: Ht = (X @ W + b)^T per batch -----------------------------
__global__ __launch_bounds__(256)
void gcn_proj(const float* __restrict__ X, const _Float16* __restrict__ Wt,
              const float* __restrict__ bias, _Float16* __restrict__ Ht) {
    __shared__ _Float16 As[TM * AS_STRIDE];
    __shared__ __align__(16) _Float16 Bt[CC * BT_STRIDE];
    const int tid  = threadIdx.x;
    const int lane = tid & 31, wave = tid >> 5;
    const int rowL = tid >> 3;   // 0..31
    const int colg = tid & 7;    // 0..7
    const size_t rowbase = (size_t)blockIdx.x * TM;   // flattened B*N rows

    v8f acc[2][4] = {};

    for (int k0 = 0; k0 < CC; k0 += KT) {
        __syncthreads();
        // X tile 32x32 fp32 -> f16
        float4 f = *(const float4*)(X + (rowbase + rowL) * CC + k0 + colg * 4);
        v4h h4; h4[0] = (_Float16)f.x; h4[1] = (_Float16)f.y;
                h4[2] = (_Float16)f.z; h4[3] = (_Float16)f.w;
        *(v4h*)(As + rowL * AS_STRIDE + colg * 4) = h4;
        // W slab: 512 cols x 32 k, two cols per thread, contiguous in Wt
#pragma unroll
        for (int j = 0; j < 2; ++j) {
            const int c = tid + j * 256;
            stage_bcol(Bt, c, Wt + (size_t)c * CC + k0);
        }
        __syncthreads();
        wmma_step(As, Bt, lane, wave, acc);
    }

    // epilogue: +bias, f16, store transposed: Ht[b][col][node]
    const int bi  = (int)(rowbase >> 12);        // rowbase / 4096
    const int nb0 = (int)(rowbase & 4095);
    const int n0 = lane & 15, rh = lane >> 4;
#pragma unroll
    for (int mt = 0; mt < 2; ++mt)
#pragma unroll
        for (int nt = 0; nt < 4; ++nt) {
            const int col = wave * 64 + nt * 16 + n0;
            const float bv = bias[col];
            v8h hv;
#pragma unroll
            for (int i = 0; i < 8; ++i) hv[i] = (_Float16)(acc[mt][nt][i] + bv);
            // 8 consecutive nodes -> one 16B store
            *(v8h*)(Ht + ((size_t)bi * CC + col) * NN + nb0 + mt * 16 + rh * 8) = hv;
        }
}

// ---- aggregation: out = (A @ H) / rowsum(A), degree fused ------------------
__global__ __launch_bounds__(256)
void gcn_agg(const float* __restrict__ A, const _Float16* __restrict__ Ht,
             float* __restrict__ out) {
    __shared__ _Float16 As[TM * AS_STRIDE];
    __shared__ __align__(16) _Float16 Bt[CC * BT_STRIDE];
    __shared__ float degpart[256];
    __shared__ float degs[TM];
    const int tid  = threadIdx.x;
    const int lane = tid & 31, wave = tid >> 5;
    const int rowL = tid >> 3;
    const int colg = tid & 7;
    const int bi    = blockIdx.x >> 7;     // NN/TM = 128 tiles per batch
    const int mtile = blockIdx.x & 127;
    const size_t Abase  = ((size_t)bi * NN + (size_t)mtile * TM) * NN;
    const size_t Htbase = (size_t)bi * CC * NN;

    v8f acc[2][4] = {};
    float dsum = 0.f;   // this thread's partial of rowsum(A) for row rowL

    for (int k0 = 0; k0 < NN; k0 += KT) {
        __syncthreads();
        // A tile 32x32 fp32 -> f16, accumulate degree in fp32 before rounding
        const float* ap = A + Abase + (size_t)rowL * NN + k0 + colg * 4;
        __builtin_prefetch(ap + KT, 0, 0);   // global_prefetch on the A stream
        float4 f = *(const float4*)ap;
        dsum += f.x + f.y + f.z + f.w;
        v4h h4; h4[0] = (_Float16)f.x; h4[1] = (_Float16)f.y;
                h4[2] = (_Float16)f.z; h4[3] = (_Float16)f.w;
        *(v4h*)(As + rowL * AS_STRIDE + colg * 4) = h4;
        // H slab: 512 cols x 32 nodes; Ht[b][c][k0..k0+31] is contiguous (L2-hot)
#pragma unroll
        for (int j = 0; j < 2; ++j) {
            const int c = tid + j * 256;
            stage_bcol(Bt, c, Ht + Htbase + (size_t)c * NN + k0);
        }
        __syncthreads();
        wmma_step(As, Bt, lane, wave, acc);
    }

    // reduce degrees: 8 threads per row
    degpart[tid] = dsum;
    __syncthreads();
    if (tid < TM) {
        float s = 0.f;
#pragma unroll
        for (int j = 0; j < 8; ++j) s += degpart[tid * 8 + j];
        degs[tid] = s;
    }
    __syncthreads();

    const int n0 = lane & 15, rh = lane >> 4;
#pragma unroll
    for (int mt = 0; mt < 2; ++mt)
#pragma unroll
        for (int nt = 0; nt < 4; ++nt) {
            const int col = wave * 64 + nt * 16 + n0;
#pragma unroll
            for (int i = 0; i < 8; ++i) {
                const int r = mt * 16 + rh * 8 + i;
                out[((size_t)bi * NN + (size_t)mtile * TM + r) * CC + col] =
                    acc[mt][nt][i] / degs[r];
            }
        }
}

extern "C" void kernel_launch(void* const* d_in, const int* in_sizes, int n_in,
                              void* d_out, int out_size, void* d_ws, size_t ws_size,
                              hipStream_t stream) {
    const float* X    = (const float*)d_in[0];   // [8,4096,512]
    const float* A    = (const float*)d_in[1];   // [8,4096,4096]
    const float* W    = (const float*)d_in[2];   // [512,512]
    const float* bias = (const float*)d_in[3];   // [512]
    float* out = (float*)d_out;

    _Float16* Ht = (_Float16*)d_ws;                       // 16M halves = 32 MB
    _Float16* Wt = Ht + (size_t)BB * NN * CC;             // +256K halves

    gcn_wcvt<<<(CC * CC) / 256, 256, 0, stream>>>(W, Wt);
    gcn_proj<<<(BB * NN) / TM, 256, 0, stream>>>(X, Wt, bias, Ht);
    gcn_agg <<<(BB * NN) / TM, 256, 0, stream>>>(A, Ht, out);
}
